// GlycanMPNN_88201448391396
// MI455X (gfx1250) — matrix operand order repaired and checked by reading, not compiled, not run on hardware
//
#include <hip/hip_runtime.h>
#include <hip/hip_bf16.h>
#include <math.h>

typedef __attribute__((ext_vector_type(16))) __bf16 v16bf;
typedef __attribute__((ext_vector_type(8)))  __bf16 v8bf;
typedef __attribute__((ext_vector_type(8)))  float  v8f;

#define NN 500000          // nodes
#define NB 8192            // graphs
#define HH 128             // hidden
#define FF 384             // feature dim (H*L)
#define NVOCAB 143

// ---------------------------------------------------------------------------
// WMMA helpers (CDNA5 wave32, D = A(16x32 bf16) * B(32x16 bf16) + C(16x16 f32))
// ---------------------------------------------------------------------------
__device__ __forceinline__ v8f wmma_bf16(v16bf a, v16bf b, v8f c) {
  return __builtin_amdgcn_wmma_f32_16x16x32_bf16(false, a, false, b, (short)0, c,
                                                 false, false);
}

// A fragment, row-major source (LDS or global), row stride ld (elements).
// Layout (ISA 7.12.2, 16-bit A 16x32): lane m   : K=0..7 (v0-3), K=16..23 (v4-7)
//                                      lane m+16: K=8..15,       K=24..31
__device__ __forceinline__ v16bf load_a(const __bf16* base, int lane, int rowBase,
                                        int kBase, int ld) {
  const int row  = rowBase + (lane & 15);
  const int koff = (lane >> 4) * 8;
  const __bf16* p = base + (size_t)row * ld + kBase + koff;
  v8bf lo = *(const v8bf*)(p);
  v8bf hi = *(const v8bf*)(p + 16);
  return __builtin_shufflevector(lo, hi, 0,1,2,3,4,5,6,7,8,9,10,11,12,13,14,15);
}

// B fragment from row-major weight W[out][in]; computes X @ W.T so
// B(k,n) = W[nBase+n][kBase+k].  Lanes 0-15: K=0..7,16..23, lanes 16-31: K=8..15,24..31,
// lane index = output column -> lane reads two 8-bf16 runs of one W row.
__device__ __forceinline__ v16bf load_b(const __bf16* w, int lane, int nBase,
                                        int kBase, int ld) {
  const int col   = lane & 15;
  const int koff  = (lane >> 4) * 8;
  const __bf16* p = w + (size_t)(nBase + col) * ld + kBase + koff;
  v8bf lo = *(const v8bf*)(p);
  v8bf hi = *(const v8bf*)(p + 16);
  return __builtin_shufflevector(lo, hi, 0,1,2,3,4,5,6,7,8,9,10,11,12,13,14,15);
}

__device__ __forceinline__ v8f bc8(float x) { v8f v = {x,x,x,x,x,x,x,x}; return v; }
__device__ __forceinline__ float sigm(float x) { return 1.0f / (1.0f + expf(-x)); }

// ---------------------------------------------------------------------------
// Prep kernels
// ---------------------------------------------------------------------------
__global__ void f32_to_bf16_kernel(const float* __restrict__ s,
                                   __bf16* __restrict__ d, int n) {
  int i = blockIdx.x * blockDim.x + threadIdx.x;
  int stride = gridDim.x * blockDim.x;
  for (; i < n; i += stride) d[i] = (__bf16)s[i];
}

// W1s[o][k] = W1[o][k] + W1[o][k+128]   (x2 = [x,x] fold), 128x128 bf16
__global__ void fold_w1_kernel(const float* __restrict__ s, __bf16* __restrict__ d) {
  int i = blockIdx.x * blockDim.x + threadIdx.x;
  if (i < HH * HH) {
    int o = i >> 7, k = i & (HH - 1);
    d[i] = (__bf16)(s[o * 2 * HH + k] + s[o * 2 * HH + HH + k]);
  }
}

__global__ void zero_kernel(unsigned int* __restrict__ p, size_t n) {
  size_t i = (size_t)blockIdx.x * blockDim.x + threadIdx.x;
  size_t stride = (size_t)gridDim.x * blockDim.x;
  for (; i < n; i += stride) p[i] = 0u;
}

// ---------------------------------------------------------------------------
// Node update stack: embed gather -> 3x (msg MLP + GRU) -> node_feature [N,384]
// One wave per 16-node tile; everything stays in LDS/registers across layers.
// ---------------------------------------------------------------------------
__global__ __launch_bounds__(32) void node_kernel(
    const int* __restrict__ unit_type, const __bf16* __restrict__ embed_bf,
    const __bf16* __restrict__ w1s, const __bf16* __restrict__ w2,
    const __bf16* __restrict__ wih, const __bf16* __restrict__ whh,
    const float* __restrict__ b1, const float* __restrict__ b2,
    const float* __restrict__ bih, const float* __restrict__ bhh,
    float* __restrict__ nf) {
  __shared__ __bf16 bufA[16 * HH];   // layer input x == bf16(h)
  __shared__ __bf16 bufB[16 * HH];   // H1 temp, then h_new (next layer input)
  __shared__ __bf16 mbf[16 * HH];    // msg (bf16)
  __shared__ float  h32[16 * HH];    // fp32 GRU state

  const int lane = threadIdx.x;
  const int rowBase = blockIdx.x * 16;
  const int n = lane & 15;
  const int mOff = (lane >> 4) * 8;

  // gather embedding rows
  for (int idx = lane; idx < 16 * HH; idx += 32) {
    int m = idx >> 7, k = idx & (HH - 1);
    __bf16 v = embed_bf[unit_type[rowBase + m] * HH + k];
    bufA[idx] = v;
    h32[idx] = (float)v;
  }
  __syncthreads();

  __bf16* xcur = bufA;
  __bf16* xnxt = bufB;

  for (int layer = 0; layer < 3; ++layer) {
    // H1 = relu(x @ W1s.T + b1) -> xnxt
    for (int nt = 0; nt < 8; ++nt) {
      v8f acc = bc8(b1[nt * 16 + n]);
      #pragma unroll
      for (int kt = 0; kt < 4; ++kt)
        acc = wmma_bf16(load_a(xcur, lane, 0, kt * 32, HH),
                        load_b(w1s, lane, nt * 16, kt * 32, HH), acc);
      #pragma unroll
      for (int r = 0; r < 8; ++r)
        xnxt[(mOff + r) * HH + nt * 16 + n] = (__bf16)fmaxf(acc[r], 0.0f);
    }
    __syncthreads();

    // msg = H1 @ W2.T + b2 -> mbf
    for (int nt = 0; nt < 8; ++nt) {
      v8f acc = bc8(b2[nt * 16 + n]);
      #pragma unroll
      for (int kt = 0; kt < 4; ++kt)
        acc = wmma_bf16(load_a(xnxt, lane, 0, kt * 32, HH),
                        load_b(w2, lane, nt * 16, kt * 32, HH), acc);
      #pragma unroll
      for (int r = 0; r < 8; ++r)
        mbf[(mOff + r) * HH + nt * 16 + n] = (__bf16)acc[r];
    }
    __syncthreads();

    // GRU: gi = msg@Wih.T+bih ; gh = h@Whh.T+bhh ; gate chunks of 16 cols
    for (int j = 0; j < 8; ++j) {
      const int col = j * 16 + n;
      v8f ir = bc8(bih[col]),        iz = bc8(bih[HH + col]),
          in_ = bc8(bih[2 * HH + col]);
      v8f hr = bc8(bhh[col]),        hz = bc8(bhh[HH + col]),
          hn_ = bc8(bhh[2 * HH + col]);
      #pragma unroll
      for (int kt = 0; kt < 4; ++kt) {
        v16bf am = load_a(mbf, lane, 0, kt * 32, HH);
        v16bf ah = load_a(xcur, lane, 0, kt * 32, HH);
        ir  = wmma_bf16(am, load_b(wih, lane,          j * 16, kt * 32, HH), ir);
        iz  = wmma_bf16(am, load_b(wih, lane, HH     + j * 16, kt * 32, HH), iz);
        in_ = wmma_bf16(am, load_b(wih, lane, 2 * HH + j * 16, kt * 32, HH), in_);
        hr  = wmma_bf16(ah, load_b(whh, lane,          j * 16, kt * 32, HH), hr);
        hz  = wmma_bf16(ah, load_b(whh, lane, HH     + j * 16, kt * 32, HH), hz);
        hn_ = wmma_bf16(ah, load_b(whh, lane, 2 * HH + j * 16, kt * 32, HH), hn_);
      }
      #pragma unroll
      for (int r = 0; r < 8; ++r) {
        const int li = (mOff + r) * HH + col;
        float ho = h32[li];
        float rg = sigm(ir[r] + hr[r]);
        float zg = sigm(iz[r] + hz[r]);
        float ng = tanhf(in_[r] + rg * hn_[r]);
        float hv = (1.0f - zg) * ng + zg * ho;
        h32[li] = hv;
        xnxt[li] = (__bf16)hv;
        nf[(size_t)(rowBase + mOff + r) * FF + layer * HH + col] = hv;
      }
    }
    __syncthreads();
    __bf16* t = xcur; xcur = xnxt; xnxt = t;
  }
}

// ---------------------------------------------------------------------------
// LSTM step: g = qs@Wih.T + bih + h@Whh.T + bhh ; gates i,f,g,o
// One wave per 16-graph tile; 4 gate accumulators in registers.
// ---------------------------------------------------------------------------
__global__ __launch_bounds__(32) void lstm_kernel(
    const __bf16* __restrict__ qs_bf, const __bf16* __restrict__ h_in_bf,
    const __bf16* __restrict__ lwih, const __bf16* __restrict__ lwhh,
    const float* __restrict__ lbih, const float* __restrict__ lbhh,
    float* __restrict__ c, float* __restrict__ h_f32,
    __bf16* __restrict__ h_out_bf) {
  const int lane = threadIdx.x;
  const int rowBase = blockIdx.x * 16;
  const int n = lane & 15;
  const int mOff = (lane >> 4) * 8;

  for (int j = 0; j < FF / 16; ++j) {
    const int col = j * 16 + n;
    v8f ai = bc8(lbih[col]          + lbhh[col]);
    v8f af = bc8(lbih[FF + col]     + lbhh[FF + col]);
    v8f ag = bc8(lbih[2 * FF + col] + lbhh[2 * FF + col]);
    v8f ao = bc8(lbih[3 * FF + col] + lbhh[3 * FF + col]);
    for (int kt = 0; kt < (2 * FF) / 32; ++kt) {   // q_star part, K = 768
      v16bf a = load_a(qs_bf, lane, rowBase, kt * 32, 2 * FF);
      ai = wmma_bf16(a, load_b(lwih, lane,          j * 16, kt * 32, 2 * FF), ai);
      af = wmma_bf16(a, load_b(lwih, lane, FF     + j * 16, kt * 32, 2 * FF), af);
      ag = wmma_bf16(a, load_b(lwih, lane, 2 * FF + j * 16, kt * 32, 2 * FF), ag);
      ao = wmma_bf16(a, load_b(lwih, lane, 3 * FF + j * 16, kt * 32, 2 * FF), ao);
    }
    for (int kt = 0; kt < FF / 32; ++kt) {         // h part, K = 384
      v16bf a = load_a(h_in_bf, lane, rowBase, kt * 32, FF);
      ai = wmma_bf16(a, load_b(lwhh, lane,          j * 16, kt * 32, FF), ai);
      af = wmma_bf16(a, load_b(lwhh, lane, FF     + j * 16, kt * 32, FF), af);
      ag = wmma_bf16(a, load_b(lwhh, lane, 2 * FF + j * 16, kt * 32, FF), ag);
      ao = wmma_bf16(a, load_b(lwhh, lane, 3 * FF + j * 16, kt * 32, FF), ao);
    }
    #pragma unroll
    for (int r = 0; r < 8; ++r) {
      const size_t gi = (size_t)(rowBase + mOff + r) * FF + col;
      float co = c[gi];
      float i_ = sigm(ai[r]), f_ = sigm(af[r]);
      float g_ = tanhf(ag[r]), o_ = sigm(ao[r]);
      float cn = f_ * co + i_ * g_;
      float hn = o_ * tanhf(cn);
      c[gi] = cn;
      h_f32[gi] = hn;
      h_out_bf[gi] = (__bf16)hn;
    }
  }
}

// ---------------------------------------------------------------------------
// Attention + readout: one workgroup per graph (<=62 nodes). Caches the
// graph's node_feature rows in LDS during the logits pass, then softmax and
// pooled sum from LDS; writes q_star (fp32 + bf16) = [h, pooled].
// ---------------------------------------------------------------------------
__global__ __launch_bounds__(256) void att_kernel(
    const float* __restrict__ nf, const float* __restrict__ h_f32,
    float* __restrict__ qs_f32, __bf16* __restrict__ qs_bf) {
  __shared__ float nfc[62 * FF];
  __shared__ float qg[FF];
  __shared__ float lg[64];
  __shared__ float att[64];
  __shared__ float red[2];

  const int tid = threadIdx.x;
  const long long g = blockIdx.x;
  const long long start = (g * (long long)NN + NB - 1) / NB;
  const long long end   = ((g + 1) * (long long)NN + NB - 1) / NB;
  const int count = (int)(end - start);

  for (int t = tid; t < FF; t += 256) qg[t] = h_f32[g * FF + t];
  __syncthreads();

  // logits + LDS cache of node rows (one wave per node)
  const int w = tid >> 5, lane = tid & 31;
  for (int i = w; i < count; i += 8) {
    const float* row = nf + (size_t)(start + i) * FF;
    float p = 0.0f;
    for (int cix = lane; cix < FF; cix += 32) {
      float v = row[cix];
      nfc[i * FF + cix] = v;
      p += v * qg[cix];
    }
    #pragma unroll
    for (int off = 16; off >= 1; off >>= 1) p += __shfl_xor(p, off, 32);
    if (lane == 0) lg[i] = p;
  }
  __syncthreads();

  if (tid == 0) {
    float m = -3.4e38f;
    for (int i = 0; i < count; ++i) m = fmaxf(m, lg[i]);
    red[0] = m;
  }
  __syncthreads();
  if (tid < 64) att[tid] = (tid < count) ? expf(lg[tid] - red[0]) : 0.0f;
  __syncthreads();
  if (tid == 0) {
    float z = 0.0f;
    for (int i = 0; i < count; ++i) z += att[i];
    red[1] = z;
  }
  __syncthreads();

  const float zinv = 1.0f / red[1];
  for (int cix = tid; cix < FF; cix += 256) {
    float s = 0.0f;
    for (int i = 0; i < count; ++i) s += att[i] * nfc[i * FF + cix];
    s *= zinv;
    qs_f32[g * 2 * FF + FF + cix] = s;
    qs_bf[g * 2 * FF + FF + cix] = (__bf16)s;
    float qv = qg[cix];
    qs_f32[g * 2 * FF + cix] = qv;
    qs_bf[g * 2 * FF + cix] = (__bf16)qv;
  }
}

// ---------------------------------------------------------------------------
// Launch
// ---------------------------------------------------------------------------
extern "C" void kernel_launch(void* const* d_in, const int* in_sizes, int n_in,
                              void* d_out, int out_size, void* d_ws, size_t ws_size,
                              hipStream_t stream) {
  const int*   unit_type = (const int*)d_in[0];
  const float* embed     = (const float*)d_in[2];
  const float* msg_W1    = (const float*)d_in[3];
  const float* msg_b1    = (const float*)d_in[4];
  const float* msg_W2    = (const float*)d_in[5];
  const float* msg_b2    = (const float*)d_in[6];
  const float* gru_Wih   = (const float*)d_in[7];
  const float* gru_Whh   = (const float*)d_in[8];
  const float* gru_bih   = (const float*)d_in[9];
  const float* gru_bhh   = (const float*)d_in[10];
  const float* lstm_Wih  = (const float*)d_in[11];
  const float* lstm_Whh  = (const float*)d_in[12];
  const float* lstm_bih  = (const float*)d_in[13];
  const float* lstm_bhh  = (const float*)d_in[14];

  float* out = (float*)d_out;
  float* nf  = out + (size_t)NB * 2 * FF;   // node_feature region of d_out

  // workspace carve-up (256B aligned)
  char* ws = (char*)d_ws;
  size_t off = 0;
  auto take = [&](size_t bytes) { char* p = ws + off; off += (bytes + 255) & ~(size_t)255; return p; };
  __bf16* embed_bf = (__bf16*)take((size_t)NVOCAB * HH * 2);
  __bf16* w1s_bf   = (__bf16*)take((size_t)HH * HH * 2);
  __bf16* w2_bf    = (__bf16*)take((size_t)HH * HH * 2);
  __bf16* wih_bf   = (__bf16*)take((size_t)3 * HH * HH * 2);
  __bf16* whh_bf   = (__bf16*)take((size_t)3 * HH * HH * 2);
  __bf16* lwih_bf  = (__bf16*)take((size_t)4 * FF * 2 * FF * 2);
  __bf16* lwhh_bf  = (__bf16*)take((size_t)4 * FF * FF * 2);
  __bf16* hbf0     = (__bf16*)take((size_t)NB * FF * 2);
  __bf16* hbf1     = (__bf16*)take((size_t)NB * FF * 2);
  float*  hf32     = (float*)take((size_t)NB * FF * 4);
  float*  cbuf     = (float*)take((size_t)NB * FF * 4);
  float*  qsf32    = (float*)take((size_t)NB * 2 * FF * 4);
  __bf16* qsbf     = (__bf16*)take((size_t)NB * 2 * FF * 2);

  // weight / embedding conversion to bf16 (deterministic, every call)
  auto cvt = [&](const float* s, __bf16* d, int n) {
    int blocks = (n + 255) / 256; if (blocks > 4096) blocks = 4096;
    f32_to_bf16_kernel<<<blocks, 256, 0, stream>>>(s, d, n);
  };
  cvt(embed,    embed_bf, NVOCAB * HH);
  cvt(msg_W2,   w2_bf,    HH * HH);
  cvt(gru_Wih,  wih_bf,   3 * HH * HH);
  cvt(gru_Whh,  whh_bf,   3 * HH * HH);
  cvt(lstm_Wih, lwih_bf,  4 * FF * 2 * FF);
  cvt(lstm_Whh, lwhh_bf,  4 * FF * FF);
  fold_w1_kernel<<<(HH * HH + 255) / 256, 256, 0, stream>>>(msg_W1, w1s_bf);

  // zero initial LSTM state: q_star(bf16), h(bf16), c(f32)
  zero_kernel<<<2048, 256, 0, stream>>>((unsigned int*)qsbf, (size_t)NB * 2 * FF / 2);
  zero_kernel<<<2048, 256, 0, stream>>>((unsigned int*)hbf0, (size_t)NB * FF / 2);
  zero_kernel<<<2048, 256, 0, stream>>>((unsigned int*)cbuf, (size_t)NB * FF);

  // node feature stack (writes nf region of d_out)
  node_kernel<<<NN / 16, 32, 0, stream>>>(unit_type, embed_bf, w1s_bf, w2_bf,
                                          wih_bf, whh_bf, msg_b1, msg_b2,
                                          gru_bih, gru_bhh, nf);

  // Set2Set: 3 sequential (LSTM -> attention) steps; h bf16 double-buffered
  __bf16* hin = hbf0;
  __bf16* hout = hbf1;
  for (int s = 0; s < 3; ++s) {
    lstm_kernel<<<NB / 16, 32, 0, stream>>>(qsbf, hin, lwih_bf, lwhh_bf,
                                            lstm_bih, lstm_bhh, cbuf, hf32, hout);
    att_kernel<<<NB, 256, 0, stream>>>(nf, hf32, qsf32, qsbf);
    __bf16* t = hin; hin = hout; hout = t;
  }

  // final q_star -> front of d_out
  hipMemcpyAsync(d_out, qsf32, (size_t)NB * 2 * FF * sizeof(float),
                 hipMemcpyDeviceToDevice, stream);
}
